// Decoder_56770877718775
// MI455X (gfx1250) — compile-verified, matrix operand order checked by reading, not run on hardware
//
#include <hip/hip_runtime.h>

// ---------------------------------------------------------------------------
// LSTM captioning decoder for MI455X (gfx1250, wave32, WMMA).
// GEMMs run on v_wmma_f32_16x16x32_f16 with f32 accumulation; tiles are
// staged into LDS with GLOBAL_LOAD_ASYNC_TO_LDS_B128 (ASYNCcnt) when the
// toolchain exposes the builtin, double-buffered across K-steps.
// ---------------------------------------------------------------------------

typedef __attribute__((ext_vector_type(16))) _Float16 v16h;
typedef __attribute__((ext_vector_type(8)))  float    v8f;
typedef __attribute__((ext_vector_type(4)))  int      v4i;

#define VOCAB 30000
#define EMB   512
#define HID   1024
#define FEAT  512
#define BATCH 256
#define TT    10
#define KDIM  1024   // FEAT+EMB == HID
#define G4H   4096   // 4*HID

// GEMM tiling
#define BM 128
#define BN 128
#define BK 32
#define LDT 40       // padded LDS stride (halves) to spread banks

// ---- async global->LDS staging (CDNA5), with safe synchronous fallback ----
#if defined(__AMDGCN__) && __has_builtin(__builtin_amdgcn_global_load_async_to_lds_b128)
#define HAVE_ASYNC_LDS 1
#else
#define HAVE_ASYNC_LDS 0
#endif

#if HAVE_ASYNC_LDS
typedef __attribute__((address_space(1))) v4i* gv4i_p;   // global int4*
typedef __attribute__((address_space(3))) v4i* lv4i_p;   // LDS int4*
#endif

__device__ __forceinline__ void cp16_to_lds(_Float16* dst, const _Float16* src)
{
#if HAVE_ASYNC_LDS
    __builtin_amdgcn_global_load_async_to_lds_b128(
        (gv4i_p)(src), (lv4i_p)(dst), 0, 0);
#else
    *(uint4*)dst = *(const uint4*)src;
#endif
}

#if HAVE_ASYNC_LDS
# if __has_builtin(__builtin_amdgcn_s_wait_asynccnt)
#  define WAIT_ASYNC_0() __builtin_amdgcn_s_wait_asynccnt(0)
#  define WAIT_ASYNC_4() __builtin_amdgcn_s_wait_asynccnt(4)
# else
#  define WAIT_ASYNC_0() asm volatile("s_wait_asynccnt 0x0" ::: "memory")
#  define WAIT_ASYNC_4() asm volatile("s_wait_asynccnt 0x4" ::: "memory")
# endif
#else
# define WAIT_ASYNC_0() ((void)0)
# define WAIT_ASYNC_4() ((void)0)
#endif

// ---------------------------------------------------------------------------
// Generic tiled WMMA GEMM:
//   out[M,N] = A[M,K] * Bw[N,K]^T  (+ bias0[n] + bias1[n] + resid[m,n])
// A, Bw are f16 row-major with leading dim K. 256 threads = 8 waves.
// Wave grid 4x2; each wave computes a 32x64 tile = 2x4 fragments of 16x16.
// Double-buffered LDS staging, 4 async b128 copies per thread per K-step.
// ---------------------------------------------------------------------------
__global__ __launch_bounds__(256)
void gemm_wmma_f16(const _Float16* __restrict__ A,
                   const _Float16* __restrict__ Bw,
                   float* __restrict__ out,
                   int M, int N, int K,
                   const float* __restrict__ bias0,
                   const float* __restrict__ bias1,
                   const float* __restrict__ resid,
                   int resid_stride)
{
    __shared__ alignas(16) _Float16 As[2][BM * LDT];
    __shared__ alignas(16) _Float16 Bs[2][BN * LDT];

    const int tid   = threadIdx.x;
    const int lane  = tid & 31;
    const int wave  = tid >> 5;
    const int waveM = wave >> 1;                 // 0..3  (x32 rows)
    const int waveN = wave & 1;                  // 0..1  (x64 cols)
    const int blockN = blockIdx.x * BN;
    const int blockM = blockIdx.y * BM;

    // per-lane fragment addressing (ISA 16-bit A/B layout):
    // lanes 0-15 hold K in {0..7, 16..23}; lanes 16-31 hold K in {8..15, 24..31}
    const int row16 = lane & 15;
    const int grp   = (lane >> 4) << 3;          // 0 or 8

    v8f acc[2][4];
    for (int i = 0; i < 2; ++i)
        for (int j = 0; j < 4; ++j)
            acc[i][j] = (v8f){0.f,0.f,0.f,0.f,0.f,0.f,0.f,0.f};

    // staging assignment: each thread copies 16 halves (2 x b128) of A and B
    const int r    = tid >> 1;                   // tile row 0..127
    const int cseg = (tid & 1) << 4;             // half-offset 0 or 16

    int ar = blockM + r; if (ar >= M) ar = M - 1;   // clamp (N tail / safety)
    int br = blockN + r; if (br >= N) br = N - 1;
    const _Float16* gA = A  + (size_t)ar * K + cseg;
    const _Float16* gB = Bw + (size_t)br * K + cseg;
    _Float16* dA[2] = { &As[0][r * LDT + cseg], &As[1][r * LDT + cseg] };
    _Float16* dB[2] = { &Bs[0][r * LDT + cseg], &Bs[1][r * LDT + cseg] };

    const int nsteps = K / BK;

    auto issue = [&](int s) {
        const int b = s & 1;
        const _Float16* a = gA + s * BK;
        const _Float16* w = gB + s * BK;
        cp16_to_lds(dA[b],     a);
        cp16_to_lds(dA[b] + 8, a + 8);
        cp16_to_lds(dB[b],     w);
        cp16_to_lds(dB[b] + 8, w + 8);
    };

    issue(0);
    for (int s = 0; s < nsteps; ++s) {
        if (s + 1 < nsteps) { issue(s + 1); WAIT_ASYNC_4(); }
        else                { WAIT_ASYNC_0(); }
        __syncthreads();

        const _Float16* sa = &As[s & 1][0];
        const _Float16* sb = &Bs[s & 1][0];

        union { v16h v; uint4 q[2]; } fa[2], fb[4];
        for (int i = 0; i < 2; ++i) {
            const _Float16* p = sa + (waveM * 32 + i * 16 + row16) * LDT;
            fa[i].q[0] = *(const uint4*)(p + grp);
            fa[i].q[1] = *(const uint4*)(p + 16 + grp);
        }
        for (int j = 0; j < 4; ++j) {
            const _Float16* p = sb + (waveN * 64 + j * 16 + row16) * LDT;
            fb[j].q[0] = *(const uint4*)(p + grp);
            fb[j].q[1] = *(const uint4*)(p + 16 + grp);
        }
        for (int i = 0; i < 2; ++i)
            for (int j = 0; j < 4; ++j)
                acc[i][j] = __builtin_amdgcn_wmma_f32_16x16x32_f16(
                    false, fa[i].v, false, fb[j].v,
                    (short)0, acc[i][j], false, false);
        __syncthreads();
    }

    // epilogue: C/D layout -> lane l: N = l%16, M = vgpr + 8*(l>=16)
    const int mrow = (lane >> 4) << 3;
    const int ncol = lane & 15;
    for (int j = 0; j < 4; ++j) {
        const int gn = blockN + waveN * 64 + j * 16 + ncol;
        if (gn >= N) continue;
        float badd = 0.f;
        if (bias0) badd += bias0[gn];
        if (bias1) badd += bias1[gn];
        for (int i = 0; i < 2; ++i) {
            const int gmBase = blockM + waveM * 32 + i * 16 + mrow;
            for (int rr = 0; rr < 8; ++rr) {
                const int gm = gmBase + rr;
                if (gm >= M) continue;
                float v = acc[i][j][rr] + badd;
                if (resid) v += resid[(size_t)gm * resid_stride + gn];
                out[(size_t)gm * N + gn] = v;
            }
        }
    }
}

// ---------------------------------------------------------------------------
// Helper kernels
// ---------------------------------------------------------------------------
__global__ void convert_f32_to_f16(const float* __restrict__ in,
                                   _Float16* __restrict__ outp, int n)
{
    int i = blockIdx.x * blockDim.x + threadIdx.x;
    if (i < n) outp[i] = (_Float16)in[i];
}

// x[m, 0:512] = features[m, :], x[m, 512:1024] = emb_table[captions[m], :]
__global__ void prep_embed(const float* __restrict__ feat,
                           const int* __restrict__ caps,
                           const float* __restrict__ emb,
                           _Float16* __restrict__ x)
{
    const int m   = blockIdx.x;           // 0..B*T-1
    const int tid = threadIdx.x;          // 256 threads, 4 cols each
    const int cap = caps[m];
    for (int c = tid * 4; c < tid * 4 + 4; ++c) {
        float v = (c < FEAT) ? feat[(size_t)m * FEAT + c]
                             : emb[(size_t)cap * EMB + (c - FEAT)];
        x[(size_t)m * KDIM + c] = (_Float16)v;
    }
}

__global__ void zero_state(float* __restrict__ c, _Float16* __restrict__ h)
{
    int i = blockIdx.x * blockDim.x + threadIdx.x;
    if (i < BATCH * HID) { c[i] = 0.f; h[i] = (_Float16)0.f; }
}

__device__ __forceinline__ float sigf(float x) { return 1.f / (1.f + __expf(-x)); }

// gates[m, 0:4H] in torch order i,f,g,o -> update c, emit h (f16) + hs row
__global__ void lstm_pointwise(const float* __restrict__ gates,
                               float* __restrict__ c,
                               _Float16* __restrict__ h16,
                               _Float16* __restrict__ hs,
                               int t)
{
    int i = blockIdx.x * blockDim.x + threadIdx.x;
    if (i >= BATCH * HID) return;
    int m = i / HID, j = i - m * HID;
    const float* g = gates + (size_t)m * G4H;
    float gi = g[j], gf = g[HID + j], gg = g[2 * HID + j], go = g[3 * HID + j];
    float cn = sigf(gf) * c[i] + sigf(gi) * tanhf(gg);
    float h  = sigf(go) * tanhf(cn);
    c[i] = cn;
    h16[i] = (_Float16)h;
    hs[((size_t)m * TT + t) * HID + j] = (_Float16)h;
}

// ---------------------------------------------------------------------------
// Launcher
// ---------------------------------------------------------------------------
extern "C" void kernel_launch(void* const* d_in, const int* in_sizes, int n_in,
                              void* d_out, int out_size, void* d_ws, size_t ws_size,
                              hipStream_t stream)
{
    const float* features = (const float*)d_in[0];   // [B,T,FEAT]
    const int*   captions = (const int*)d_in[1];     // [B,T]
    const float* emb      = (const float*)d_in[2];   // [VOCAB,EMB]
    const float* W_ih     = (const float*)d_in[3];   // [4H, FEAT+EMB]
    const float* W_hh     = (const float*)d_in[4];   // [4H, HID]
    const float* b_ih     = (const float*)d_in[5];   // [4H]
    const float* b_hh     = (const float*)d_in[6];   // [4H]
    const float* fc_W     = (const float*)d_in[7];   // [VOCAB, HID]
    const float* fc_b     = (const float*)d_in[8];   // [VOCAB]
    float* out = (float*)d_out;                      // [B,T,VOCAB]

    // workspace carve-up (all sub-buffers multiple of 256B)
    char* p = (char*)d_ws;
    _Float16* x16   = (_Float16*)p; p += (size_t)BATCH * TT * KDIM * 2;  //  5.2 MB
    _Float16* Wih16 = (_Float16*)p; p += (size_t)G4H * KDIM * 2;         //  8.4 MB
    _Float16* Whh16 = (_Float16*)p; p += (size_t)G4H * HID * 2;         //  8.4 MB
    _Float16* fcW16 = (_Float16*)p; p += (size_t)VOCAB * HID * 2;        // 61.4 MB
    float*    gx    = (float*)p;    p += (size_t)BATCH * TT * G4H * 4;   // 41.9 MB
    float*    gates = (float*)p;    p += (size_t)BATCH * G4H * 4;        //  4.2 MB
    float*    cst   = (float*)p;    p += (size_t)BATCH * HID * 4;        //  1.0 MB
    _Float16* h16   = (_Float16*)p; p += (size_t)BATCH * HID * 2;        //  0.5 MB
    _Float16* hs16  = (_Float16*)p; p += (size_t)BATCH * TT * HID * 2;   //  5.2 MB

    // 1) per-call weight conversion to f16 (no cross-call caching allowed)
    { int n = G4H * KDIM;  convert_f32_to_f16<<<(n + 255) / 256, 256, 0, stream>>>(W_ih, Wih16, n); }
    { int n = G4H * HID;   convert_f32_to_f16<<<(n + 255) / 256, 256, 0, stream>>>(W_hh, Whh16, n); }
    { int n = VOCAB * HID; convert_f32_to_f16<<<(n + 255) / 256, 256, 0, stream>>>(fc_W, fcW16, n); }

    // 2) embedding gather + concat -> f16 input matrix [B*T, 1024]
    prep_embed<<<BATCH * TT, 256, 0, stream>>>(features, captions, emb, x16);

    // 3) zero-init recurrent state
    { int n = BATCH * HID; zero_state<<<(n + 255) / 256, 256, 0, stream>>>(cst, h16); }

    // 4) gx = x @ W_ih^T + b_ih + b_hh   [2560 x 4096]
    {
        dim3 grid(G4H / BN, (BATCH * TT) / BM);
        gemm_wmma_f16<<<grid, 256, 0, stream>>>(x16, Wih16, gx,
            BATCH * TT, G4H, KDIM, b_ih, b_hh, nullptr, 0);
    }

    // 5) recurrence: gates = h @ W_hh^T + gx[:,t,:]; then pointwise update
    {
        dim3 grid(G4H / BN, BATCH / BM);
        for (int t = 0; t < TT; ++t) {
            gemm_wmma_f16<<<grid, 256, 0, stream>>>(h16, Whh16, gates,
                BATCH, G4H, HID, nullptr, nullptr,
                gx + (size_t)t * G4H, TT * G4H);
            int n = BATCH * HID;
            lstm_pointwise<<<(n + 255) / 256, 256, 0, stream>>>(gates, cst, h16, hs16, t);
        }
    }

    // 6) logits = hs @ fc_W^T + fc_b   [2560 x 30000]
    {
        dim3 grid((VOCAB + BN - 1) / BN, (BATCH * TT) / BM);
        gemm_wmma_f16<<<grid, 256, 0, stream>>>(hs16, fcW16, out,
            BATCH * TT, VOCAB, HID, fc_b, nullptr, nullptr, 0);
    }
}
